// MLPMoE_5282809774198
// MI455X (gfx1250) — matrix operand order repaired and checked by reading, not compiled, not run on hardware
//
#include <hip/hip_runtime.h>
#include <hip/hip_bf16.h>

typedef __bf16 bf16_t;
typedef bf16_t v16bf __attribute__((ext_vector_type(16)));
typedef bf16_t v8bf  __attribute__((ext_vector_type(8)));
typedef float  v8f   __attribute__((ext_vector_type(8)));
typedef int    v4i   __attribute__((ext_vector_type(4)));

#define B_  16
#define N_  576
#define D_  1024
#define C_  4096
#define E_  4
#define M_TOK (B_*N_)          /* 9216 tokens */
#define OUT_MAIN ((size_t)M_TOK*(size_t)C_)

/* workspace byte offsets */
#define WS_STATS   0           /* floats: [0]=sum z^2, [1..64]=d1proxy, [65..128]=d1 */
#define WS_COMBINE 4096        /* M_TOK*E_ floats = 147456 B */
#define WS_XBF     151552      /* M_TOK*D_ ushort = 18874368 B */
#define WS_HBF     19025920    /* M_TOK*C_ ushort = 75497472 B -> end 94523392 */
#define WS_W1BF    94523392    /* C_*D_ ushort = 8388608 B (per-expert reuse) */
#define WS_W2BF    102912000   /* C_*C_ ushort = 33554432 B (per-expert reuse) -> end ~136.5MB */

static __device__ __forceinline__ unsigned short f2bf(float f) {
  union { float f; unsigned u; } v; v.f = f;
  unsigned u = v.u;
  u = (u + 0x7FFFu + ((u >> 16) & 1u)) >> 16;   /* round-to-nearest-even */
  return (unsigned short)u;
}

/* ---- CDNA5 async global->LDS copy (ASYNCcnt path) ---- */
static __device__ __forceinline__ void async_copy_b128(const unsigned short* gsrc,
                                                       unsigned short* ldst) {
#if __has_builtin(__builtin_amdgcn_global_load_async_to_lds_b128)
  __builtin_amdgcn_global_load_async_to_lds_b128(
      (__attribute__((address_space(1))) v4i*)(size_t)(const void*)gsrc,
      (__attribute__((address_space(3))) v4i*)(void*)ldst, 0, 0);
#else
  unsigned lds32 = (unsigned)(size_t)(void*)ldst;     /* LDS aperture: offset in low 32 bits */
  unsigned long long g64 = (unsigned long long)(size_t)(const void*)gsrc;
  asm volatile("global_load_async_to_lds_b128 %0, %1, off" :: "v"(lds32), "v"(g64) : "memory");
#endif
}
static __device__ __forceinline__ void wait_async0() {
#if __has_builtin(__builtin_amdgcn_s_wait_asynccnt)
  __builtin_amdgcn_s_wait_asynccnt(0);
#else
  asm volatile("s_wait_asynccnt 0" ::: "memory");
#endif
}

/* ---------------- zero stats ---------------- */
__global__ void zero_stats_kernel(float* stats) {
  if (threadIdx.x < 160) stats[threadIdx.x] = 0.0f;
}

/* ---------------- f32 -> bf16 convert (X and weight slices) ---------------- */
__global__ __launch_bounds__(256) void cvt_bf16_kernel(const float* __restrict__ src,
                                                       unsigned short* __restrict__ dst) {
  size_t i = ((size_t)blockIdx.x * 256 + threadIdx.x) * 4;
  float4 v = *(const float4*)(src + i);
  union { unsigned short s[4]; uint2 u; } o;
  o.s[0] = f2bf(v.x); o.s[1] = f2bf(v.y); o.s[2] = f2bf(v.z); o.s[3] = f2bf(v.w);
  *(uint2*)(dst + i) = o.u;
}

/* ---------------- router: logits, losses, top-2 combine weights ---------------- */
__global__ __launch_bounds__(256) void router_kernel(const float* __restrict__ x,
                                                     const float* __restrict__ Wg,
                                                     float* __restrict__ combine,
                                                     float* __restrict__ stats) {
  int wid = threadIdx.x >> 5, lane = threadIdx.x & 31;
  int tok = blockIdx.x * 8 + wid;              /* grid = M_TOK/8 blocks exactly */
  const float* xr = x + (size_t)tok * D_;
  float a0 = 0.f, a1 = 0.f, a2 = 0.f, a3 = 0.f;
  for (int d = lane; d < D_; d += 32) {
    float xv = xr[d];
    a0 += xv * Wg[0 * D_ + d];
    a1 += xv * Wg[1 * D_ + d];
    a2 += xv * Wg[2 * D_ + d];
    a3 += xv * Wg[3 * D_ + d];
  }
  for (int off = 16; off; off >>= 1) {
    a0 += __shfl_xor(a0, off, 32);
    a1 += __shfl_xor(a1, off, 32);
    a2 += __shfl_xor(a2, off, 32);
    a3 += __shfl_xor(a3, off, 32);
  }
  if (lane == 0) {
    float g[4] = { a0, a1, a2, a3 };
    float mx = fmaxf(fmaxf(g[0], g[1]), fmaxf(g[2], g[3]));
    float p[4]; float s = 0.f;
    #pragma unroll
    for (int e = 0; e < 4; e++) { p[e] = __expf(g[e] - mx); s += p[e]; }
    float z = mx + __logf(s);
    atomicAdd(&stats[0], z * z);
    float inv = 1.0f / s;
    #pragma unroll
    for (int e = 0; e < 4; e++) p[e] *= inv;   /* softmax */
    int b = tok / N_;
    #pragma unroll
    for (int e = 0; e < 4; e++) atomicAdd(&stats[1 + b * E_ + e], p[e]);
    /* top-2 (ties -> lower index, like jax top_k) */
    int i0 = 0;
    #pragma unroll
    for (int e = 1; e < 4; e++) if (p[e] > p[i0]) i0 = e;
    int i1 = (i0 == 0) ? 1 : 0;
    #pragma unroll
    for (int e = 0; e < 4; e++) if (e != i0 && e != i1 && p[e] > p[i1]) i1 = e;
    atomicAdd(&stats[65 + b * E_ + i0], 1.0f);
    float wsum = p[i0] + p[i1];
    float c[4] = { 0.f, 0.f, 0.f, 0.f };
    c[i0] = p[i0] / wsum;
    c[i1] = p[i1] / wsum;
    float4 cv; cv.x = c[0]; cv.y = c[1]; cv.z = c[2]; cv.w = c[3];
    *(float4*)(combine + (size_t)tok * 4) = cv;
  }
}

/* ---------------- WMMA GEMM:  Out[M,NN] = A[M,K](bf16) * Bw[NN,K](bf16)^T ----------------
   Both operands DMA'd global->LDS via async copies; no staging registers. */
__global__ __launch_bounds__(256) void gemm_ffn_kernel(
    const unsigned short* __restrict__ A,    /* bf16 raw, row-major [M_TOK, K]      */
    const unsigned short* __restrict__ Bw,   /* bf16 raw, row-major [C_, K]         */
    const float* __restrict__ bias,          /* [C_]                                */
    int K,                                   /* 1024 (ffn1) or 4096 (ffn2)          */
    int mode,                                /* 0: gelu -> Hout ; 1: combine -> out */
    unsigned short* __restrict__ Hout,
    float* __restrict__ outF,
    const float* __restrict__ combine,
    int e, int firstExpert) {
  constexpr int BM = 128, BN = 128, BK = 32, LDT = 48; /* LDS row stride 96B (16B mult) */
  __shared__ unsigned short sA[2][BM * LDT];
  __shared__ unsigned short sB[2][BN * LDT];

  const int tid  = threadIdx.x;
  const int lane = tid & 31;
  const int wid  = tid >> 5;
  const int wm   = wid >> 2;      /* 0..1 */
  const int wn   = wid & 3;       /* 0..3 */
  const int bm   = blockIdx.y * BM;
  const int bn   = blockIdx.x * BN;
  const int KT   = K / BK;

  /* each thread DMAs 2x16B of A-tile and 2x16B of B-tile per K-step */
  auto issueTile = [&](int buf, int kt) {
    #pragma unroll
    for (int i = 0; i < 2; i++) {
      int idx = tid + i * 256;
      int row = idx >> 2, ch = idx & 3;
      async_copy_b128(A + (size_t)(bm + row) * K + (size_t)kt * BK + ch * 8,
                      &sA[buf][row * LDT + ch * 8]);
    }
    #pragma unroll
    for (int i = 0; i < 2; i++) {
      int idx = tid + i * 256;
      int row = idx >> 2, ch = idx & 3;
      async_copy_b128(Bw + (size_t)(bn + row) * K + (size_t)kt * BK + ch * 8,
                      &sB[buf][row * LDT + ch * 8]);
    }
  };

  v8f acc[4][2];
  #pragma unroll
  for (int i = 0; i < 4; i++)
    #pragma unroll
    for (int j = 0; j < 2; j++)
      acc[i][j] = (v8f){0.f, 0.f, 0.f, 0.f, 0.f, 0.f, 0.f, 0.f};

  issueTile(0, 0);
  wait_async0();
  __syncthreads();

  const int hi = lane >> 4;       /* half-wave select per ISA fragment tables */
  const int lr = lane & 15;

  for (int kt = 0; kt < KT; ++kt) {
    int cur = kt & 1;
    if (kt + 1 < KT) issueTile(cur ^ 1, kt + 1);   /* DMA next tile, overlap with WMMA */

    const unsigned short* pa = sA[cur];
    const unsigned short* pb = sB[cur];

    v16bf af[4], bfm[2];
    /* A 16x32 bf16 frag: lane<16 holds M=lr, K={0..7}+{16..23}; lane>=16: K={8..15}+{24..31} */
    #pragma unroll
    for (int i = 0; i < 4; i++) {
      int row = wm * 64 + i * 16 + lr;
      v8bf lo  = *(const v8bf*)(pa + row * LDT + hi * 8);
      v8bf hi8 = *(const v8bf*)(pa + row * LDT + hi * 8 + 16);
      af[i] = __builtin_shufflevector(lo, hi8, 0,1,2,3,4,5,6,7,8,9,10,11,12,13,14,15);
    }
    /* B 32x16 bf16 frag: lane<16 holds N=lr, K={0..7}+{16..23}; lane>=16: K={8..15}+{24..31} */
    #pragma unroll
    for (int j = 0; j < 2; j++) {
      int col = wn * 32 + j * 16 + lr;
      v8bf lo  = *(const v8bf*)(pb + col * LDT + hi * 8);
      v8bf hi8 = *(const v8bf*)(pb + col * LDT + hi * 8 + 16);
      bfm[j] = __builtin_shufflevector(lo, hi8, 0,1,2,3,4,5,6,7,8,9,10,11,12,13,14,15);
    }
    #pragma unroll
    for (int i = 0; i < 4; i++)
      #pragma unroll
      for (int j = 0; j < 2; j++)
        acc[i][j] = __builtin_amdgcn_wmma_f32_16x16x32_bf16(
            false, af[i], false, bfm[j], (short)0, acc[i][j], false, false);

    wait_async0();        /* own DMA for next tile done (after compute = overlapped) */
    __syncthreads();      /* everyone's DMA visible; cur buffer free for reuse */
  }

  /* epilogue; C/D layout: VGPR r -> M=r (lanes 0-15) / M=r+8 (lanes 16-31), N=lane&15 */
  const int mh = hi * 8;
  #pragma unroll
  for (int i = 0; i < 4; i++) {
    #pragma unroll
    for (int j = 0; j < 2; j++) {
      int col = bn + wn * 32 + j * 16 + lr;
      float bv = bias[col];
      #pragma unroll
      for (int r = 0; r < 8; r++) {
        int row = bm + wm * 64 + i * 16 + mh + r;
        float v = acc[i][j][r] + bv;
        if (mode == 0) {
          float g = 0.5f * v * (1.0f + erff(v * 0.70710678118654752f));  /* exact gelu */
          Hout[(size_t)row * C_ + col] = f2bf(g);
        } else {
          float w  = combine[(size_t)row * E_ + e];
          size_t o = (size_t)row * C_ + col;
          float val = w * v;
          if (firstExpert) outF[o] = val;
          else             outF[o] += val;
        }
      }
    }
  }
}

/* ---------------- finalize scalar losses ---------------- */
__global__ void finalize_kernel(const float* __restrict__ stats, float* __restrict__ out) {
  if (threadIdx.x == 0 && blockIdx.x == 0) {
    float s = 0.f;
    for (int i = 0; i < B_ * E_; i++) {
      float dp = stats[1 + i] / (float)N_;
      float d1 = stats[65 + i] / (float)N_;
      s += dp * d1;
    }
    out[OUT_MAIN + 0] = s / (float)(B_ * E_) * (float)(E_ * E_); /* balance loss */
    out[OUT_MAIN + 1] = stats[0] / (float)(B_ * N_);             /* router z loss */
  }
}

extern "C" void kernel_launch(void* const* d_in, const int* in_sizes, int n_in,
                              void* d_out, int out_size, void* d_ws, size_t ws_size,
                              hipStream_t stream) {
  const float* x  = (const float*)d_in[0];
  const float* Wg = (const float*)d_in[1];
  const float* W1 = (const float*)d_in[2];
  const float* b1 = (const float*)d_in[3];
  const float* W2 = (const float*)d_in[4];
  const float* b2 = (const float*)d_in[5];
  float* out = (float*)d_out;

  char* ws = (char*)d_ws;
  float*          stats   = (float*)(ws + WS_STATS);
  float*          combine = (float*)(ws + WS_COMBINE);
  unsigned short* Xb      = (unsigned short*)(ws + WS_XBF);
  unsigned short* Hb      = (unsigned short*)(ws + WS_HBF);
  unsigned short* W1b     = (unsigned short*)(ws + WS_W1BF);
  unsigned short* W2b     = (unsigned short*)(ws + WS_W2BF);

  zero_stats_kernel<<<1, 256, 0, stream>>>(stats);
  cvt_bf16_kernel<<<(M_TOK * D_) / 1024, 256, 0, stream>>>(x, Xb);
  router_kernel<<<M_TOK / 8, 256, 0, stream>>>(x, Wg, combine, stats);

  dim3 grid(C_ / 128, M_TOK / 128);
  for (int e = 0; e < E_; e++) {
    cvt_bf16_kernel<<<(C_ * D_) / 1024, 256, 0, stream>>>(W1 + (size_t)e * C_ * D_, W1b);
    gemm_ffn_kernel<<<grid, 256, 0, stream>>>(
        Xb, W1b, b1 + (size_t)e * C_, D_,
        0, Hb, nullptr, nullptr, e, 0);
    cvt_bf16_kernel<<<(C_ * C_) / 1024, 256, 0, stream>>>(W2 + (size_t)e * C_ * C_, W2b);
    gemm_ffn_kernel<<<grid, 256, 0, stream>>>(
        Hb, W2b, b2 + (size_t)e * C_, C_,
        1, nullptr, out, combine, e, (e == 0) ? 1 : 0);
  }
  finalize_kernel<<<1, 64, 0, stream>>>(stats, out);
}
